// MCI_GRU_Cell_14516989460524
// MI455X (gfx1250) — matrix operand order repaired and checked by reading, not compiled, not run on hardware
//
#include <hip/hip_runtime.h>
#include <hip/hip_bf16.h>
#include <math.h>

// ---------------------------------------------------------------------------
// GRU (attention branch is dead code in the reference) on MI455X / gfx1250.
//   * batch rows independent -> 32 persistent workgroups of 16 rows each
//   * weights converted per launch to bf16 WMMA B-fragment layout in d_ws
//   * per timestep: 7 GEMM fragments via v_wmma_f32_16x16x32_bf16, h kept in
//     accumulator-layout registers, bf16 copies of h / r*h / x_t in LDS.
//   * k-tile loops kept ROLLED (unroll_count 2) + sched barriers between
//     phases to stop the scheduler prefetching ~700 weight loads at once,
//     which previously caused scratch spills of every WMMA B operand.
// ---------------------------------------------------------------------------

typedef __attribute__((ext_vector_type(16))) __bf16 v16bf;
typedef __attribute__((ext_vector_type(8)))  __bf16 v8bf;
typedef __attribute__((ext_vector_type(8)))  float  v8f;

#define UNITS 512
#define D_IN  256
#define BATCH 512
#define TLEN  128
#define MTILE 16           // batch rows per workgroup (one WMMA M tile)
#define NT_PER_WAVE 4      // 4 n-tiles of 16 -> 64 columns per wave
#define THREADS 256        // 8 waves

// d_ws offsets in bf16 elements (fragment-packed weights)
#define OFF_WZX 0
#define OFF_WRX (OFF_WZX + D_IN*UNITS)
#define OFF_WHX (OFF_WRX + D_IN*UNITS)
#define OFF_WZH (OFF_WHX + D_IN*UNITS)
#define OFF_WRH (OFF_WZH + UNITS*UNITS)
#define OFF_WHH (OFF_WRH + UNITS*UNITS)
#define OFF_WO  (OFF_WHH + UNITS*UNITS)

#if defined(__has_builtin)
#  if __has_builtin(__builtin_amdgcn_sched_barrier)
#    define SCHED_FENCE() __builtin_amdgcn_sched_barrier(0)
#  endif
#endif
#ifndef SCHED_FENCE
#  define SCHED_FENCE() asm volatile("" ::: "memory")
#endif

union U16 { v16bf v; v8bf h[2]; };

// ---------------------------------------------------------------------------
// Reformat one f32 KxN row-major weight matrix into bf16 WMMA B-fragments.
// Fragment (kt, nt) covers K rows [kt*32, kt*32+32), N cols [nt*16, nt*16+16).
// Within a fragment (512 bf16): element index = lane*16 + e, where
//   k = kt*32 + (lane>=16 ? 8 : 0) + (e < 8 ? e : 8 + e),  n = nt*16 + (lane & 15)
// matching the CDNA5 16-bit B-matrix VGPR layout.
// ---------------------------------------------------------------------------
__global__ void reformat_weight(const float* __restrict__ src,
                                __bf16* __restrict__ dst, int K, int N) {
  int idx = blockIdx.x * blockDim.x + threadIdx.x;
  int total = K * N;
  if (idx >= total) return;
  int ktiles = K >> 5;
  int f      = idx >> 9;        // fragment index = nt*ktiles + kt
  int within = idx & 511;
  int lane   = within >> 4;
  int e      = within & 15;
  int kt = f % ktiles;
  int nt = f / ktiles;
  int k = (kt << 5) + ((lane >> 4) << 3) + (e < 8 ? e : 8 + e);
  int n = (nt << 4) + (lane & 15);
  dst[idx] = (__bf16)src[k * N + n];
}

// ---------------------------------------------------------------------------
// Accumulate a [16 x K] (LDS, bf16, row-major) times [K x 64] (fragment-packed
// global weights) into 4 accumulator tiles for this wave. Loop kept rolled
// (2x pipelined) to bound register pressure.
// ---------------------------------------------------------------------------
__device__ __forceinline__ void gemm_acc16(const __bf16* sA, int K,
                                           const __bf16* __restrict__ Bfrag,
                                           int ntBase, int lane,
                                           v8f acc[NT_PER_WAVE]) {
  const int ktiles = K >> 5;
  const int r    = lane & 15;
  const int koff = (lane >> 4) << 3;            // 0 or 8
  const __bf16* rowp = sA + r * K;
#pragma clang loop unroll_count(2)
  for (int kt = 0; kt < ktiles; ++kt) {
    U16 a;
    const __bf16* pa = rowp + (kt << 5) + koff;
    a.h[0] = *(const v8bf*)(pa);                // K = k0 .. k0+7   (or +8..15)
    a.h[1] = *(const v8bf*)(pa + 16);           // K = k0+16..k0+23 (or +24..31)
#pragma unroll
    for (int i = 0; i < NT_PER_WAVE; ++i) {
      const __bf16* pb = Bfrag + (((size_t)((ntBase + i) * ktiles + kt)) << 9)
                               + (lane << 4);
      U16 b;
      b.h[0] = *(const v8bf*)(pb);
      b.h[1] = *(const v8bf*)(pb + 8);
      acc[i] = __builtin_amdgcn_wmma_f32_16x16x32_bf16(
          false, a.v, false, b.v, (short)0, acc[i], false, false);
    }
  }
}

__device__ __forceinline__ void fill_acc(v8f acc[NT_PER_WAVE], const float b[NT_PER_WAVE]) {
#pragma unroll
  for (int i = 0; i < NT_PER_WAVE; ++i)
#pragma unroll
    for (int j = 0; j < 8; ++j) acc[i][j] = b[i];
}

// ---------------------------------------------------------------------------
// Persistent GRU recurrence: one workgroup owns 16 batch rows for all T steps.
// ---------------------------------------------------------------------------
__global__ void __launch_bounds__(THREADS, 1)
gru_recurrence(const float* __restrict__ x,
               const float* __restrict__ b_z, const float* __restrict__ b_r,
               const float* __restrict__ b_h, const float* __restrict__ b_o,
               const __bf16* __restrict__ wf,
               float* __restrict__ out) {
  __shared__ __align__(16) __bf16 s_x [MTILE * D_IN ];  //  8 KB  x_t tile
  __shared__ __align__(16) __bf16 s_hb[MTILE * UNITS];  // 16 KB  h (bf16)
  __shared__ __align__(16) __bf16 s_rh[MTILE * UNITS];  // 16 KB  r*h (bf16)

  const int tid    = threadIdx.x;
  const int lane   = tid & 31;
  const int wave   = tid >> 5;
  const int b0     = blockIdx.x * MTILE;
  const int ntBase = wave * NT_PER_WAVE;
  const int m_hi   = (lane >> 4) << 3;           // row offset 0 or 8 in C layout
  const int ncol0  = (ntBase << 4) + (lane & 15);

  for (int i = tid; i < MTILE * UNITS; i += THREADS) s_hb[i] = (__bf16)0.0f;

  v8f h_reg[NT_PER_WAVE];                        // h in accumulator layout
#pragma unroll
  for (int i = 0; i < NT_PER_WAVE; ++i)
#pragma unroll
    for (int j = 0; j < 8; ++j) h_reg[i][j] = 0.0f;

  float bz[NT_PER_WAVE], br[NT_PER_WAVE], bh[NT_PER_WAVE], bo[NT_PER_WAVE];
#pragma unroll
  for (int i = 0; i < NT_PER_WAVE; ++i) {
    int n = ncol0 + (i << 4);
    bz[i] = b_z[n]; br[i] = b_r[n]; bh[i] = b_h[n]; bo[i] = b_o[n];
  }

  for (int t = 0; t < TLEN; ++t) {
    __syncthreads();
    { // stage x_t tile -> LDS bf16 (thread i: row i/16, 16 contiguous cols)
      int row = tid >> 4;
      int c0  = (tid & 15) << 4;
      const float* px = x + ((size_t)(b0 + row) * TLEN + t) * D_IN + c0;
#pragma unroll
      for (int j = 0; j < 16; ++j) s_x[row * D_IN + c0 + j] = (__bf16)px[j];
    }
    __syncthreads();
    SCHED_FENCE();

    // ---- gates z, r ----
    v8f zacc[NT_PER_WAVE], racc[NT_PER_WAVE];
    fill_acc(zacc, bz);
    fill_acc(racc, br);
    gemm_acc16(s_x,  D_IN,  wf + OFF_WZX, ntBase, lane, zacc);
    SCHED_FENCE();
    gemm_acc16(s_hb, UNITS, wf + OFF_WZH, ntBase, lane, zacc);
    SCHED_FENCE();
    gemm_acc16(s_x,  D_IN,  wf + OFF_WRX, ntBase, lane, racc);
    SCHED_FENCE();
    gemm_acc16(s_hb, UNITS, wf + OFF_WRH, ntBase, lane, racc);
    SCHED_FENCE();

#pragma unroll
    for (int i = 0; i < NT_PER_WAVE; ++i) {
      int n = ncol0 + (i << 4);
#pragma unroll
      for (int j = 0; j < 8; ++j) {
        float zv = 1.0f / (1.0f + __expf(-zacc[i][j]));
        float rv = 1.0f / (1.0f + __expf(-racc[i][j]));
        zacc[i][j] = zv;                                // keep z in registers
        s_rh[(m_hi + j) * UNITS + n] = (__bf16)(rv * h_reg[i][j]);
      }
    }
    __syncthreads();                                    // r*h complete
    SCHED_FENCE();

    // ---- candidate + GRU update ----
    v8f hacc[NT_PER_WAVE];
    fill_acc(hacc, bh);
    gemm_acc16(s_x,  D_IN,  wf + OFF_WHX, ntBase, lane, hacc);
    SCHED_FENCE();
    gemm_acc16(s_rh, UNITS, wf + OFF_WHH, ntBase, lane, hacc);
    SCHED_FENCE();

#pragma unroll
    for (int i = 0; i < NT_PER_WAVE; ++i) {
      int n = ncol0 + (i << 4);
#pragma unroll
      for (int j = 0; j < 8; ++j) {
        float hc = tanhf(hacc[i][j]);
        float zv = zacc[i][j];
        float hn = (1.0f - zv) * h_reg[i][j] + zv * hc;
        h_reg[i][j] = hn;
        s_hb[(m_hi + j) * UNITS + n] = (__bf16)hn;
      }
    }
    __syncthreads();                                    // h_new complete
    SCHED_FENCE();

    // ---- output projection ----
    v8f oacc[NT_PER_WAVE];
    fill_acc(oacc, bo);
    gemm_acc16(s_hb, UNITS, wf + OFF_WO, ntBase, lane, oacc);
    SCHED_FENCE();
#pragma unroll
    for (int i = 0; i < NT_PER_WAVE; ++i) {
      int n = ncol0 + (i << 4);
#pragma unroll
      for (int j = 0; j < 8; ++j) {
        int m = m_hi + j;
        out[((size_t)(b0 + m) * TLEN + t) * UNITS + n] = oacc[i][j];
      }
    }
  }
}

extern "C" void kernel_launch(void* const* d_in, const int* in_sizes, int n_in,
                              void* d_out, int out_size, void* d_ws, size_t ws_size,
                              hipStream_t stream) {
  (void)in_sizes; (void)n_in; (void)out_size; (void)ws_size;
  const float* x   = (const float*)d_in[0];
  const float* Wzx = (const float*)d_in[1];
  const float* Wrx = (const float*)d_in[2];
  const float* Whx = (const float*)d_in[3];
  const float* Wzh = (const float*)d_in[4];
  const float* Wrh = (const float*)d_in[5];
  const float* Whh = (const float*)d_in[6];
  const float* bz  = (const float*)d_in[7];
  const float* br  = (const float*)d_in[8];
  const float* bh  = (const float*)d_in[9];
  // d_in[10..12] = W_q, W_k, W_v -> dead code in the reference, skipped.
  const float* Wo  = (const float*)d_in[13];
  const float* bo  = (const float*)d_in[14];
  float* out = (float*)d_out;
  __bf16* wf = (__bf16*)d_ws;

  // Repack live weights into bf16 WMMA B-fragment layout (every call; det.).
  reformat_weight<<<(D_IN*UNITS + 255)/256, 256, 0, stream>>>(Wzx, wf + OFF_WZX, D_IN,  UNITS);
  reformat_weight<<<(D_IN*UNITS + 255)/256, 256, 0, stream>>>(Wrx, wf + OFF_WRX, D_IN,  UNITS);
  reformat_weight<<<(D_IN*UNITS + 255)/256, 256, 0, stream>>>(Whx, wf + OFF_WHX, D_IN,  UNITS);
  reformat_weight<<<(UNITS*UNITS + 255)/256, 256, 0, stream>>>(Wzh, wf + OFF_WZH, UNITS, UNITS);
  reformat_weight<<<(UNITS*UNITS + 255)/256, 256, 0, stream>>>(Wrh, wf + OFF_WRH, UNITS, UNITS);
  reformat_weight<<<(UNITS*UNITS + 255)/256, 256, 0, stream>>>(Whh, wf + OFF_WHH, UNITS, UNITS);
  reformat_weight<<<(UNITS*UNITS + 255)/256, 256, 0, stream>>>(Wo,  wf + OFF_WO,  UNITS, UNITS);

  gru_recurrence<<<BATCH / MTILE, THREADS, 0, stream>>>(x, bz, br, bh, bo, wf, out);
}
